// APPNP_25357486915691
// MI455X (gfx1250) — matrix operand order repaired and checked by reading, not compiled, not run on hardware
//
#include <hip/hip_runtime.h>
#include <stdint.h>

// CDNA5 / gfx1250 WMMA fragment types (wave32)
typedef __attribute__((ext_vector_type(16))) __bf16 v16bf;
typedef __attribute__((ext_vector_type(8)))  float  v8f;

// fp32 -> bf16, round-half-up (1 VALU). Tie behavior differs from RNE by <=1ulp.
__device__ __forceinline__ unsigned short f2bf(float f) {
  return (unsigned short)((__float_as_uint(f) + 0x8000u) >> 16);
}

// pack two fp32 -> two bf16 in one dword: {bf(hi), bf(lo)} via v_perm_b32
__device__ __forceinline__ unsigned pk2bf(float lo, float hi) {
  unsigned a = __float_as_uint(lo) + 0x8000u;
  unsigned b = __float_as_uint(hi) + 0x8000u;
  // result bytes {0,1} = a bytes {2,3}; bytes {2,3} = b bytes {2,3}
  return __builtin_amdgcn_perm(b, a, 0x07060302u);
}

union AB32 {
  v16bf v;
  uint4 q[2];
  unsigned d[8];
  unsigned short s[16];
};

// ---------------------------------------------------------------------------
// Weight convert + transpose: src fp32 [K][Nc] row-major -> dst bf16 [Nc][K]
// ---------------------------------------------------------------------------
__global__ void __launch_bounds__(256) k_cvt_t(const float* __restrict__ src,
                                               unsigned short* __restrict__ dst,
                                               int K, int Nc) {
  int idx = blockIdx.x * blockDim.x + threadIdx.x;
  if (idx >= K * Nc) return;
  int k = idx / Nc;
  int n = idx - k * Nc;
  dst[n * K + k] = f2bf(src[idx]);
}

// ---------------------------------------------------------------------------
// Fused MLP: out += alpha * ( relu(X @ W0 + b0) @ W1 + b1 )
// X: [N][Cin] fp32, W0T: [256][Cin] bf16 (transposed), W1T: [64][256] bf16.
// Block = 256 threads = 8 waves; 64 rows/block.
// Wave w: row-tile rt = w&3 (16 rows), col-half ch = w>>2 (128 of 256 hidden).
// ---------------------------------------------------------------------------
__global__ void __launch_bounds__(256) k_mlp(const float* __restrict__ X,
    const unsigned short* __restrict__ W0T, const float* __restrict__ b0,
    const unsigned short* __restrict__ W1T, const float* __restrict__ b1,
    float* __restrict__ out, int N, int Cin, float alpha, int accum) {
  extern __shared__ char smem[];
  unsigned short* ldsB = (unsigned short*)smem;            // 256 x 32 bf16 = 16KB
  unsigned short* ldsH = (unsigned short*)(smem + 16384);  // 64 x 256 bf16 = 32KB

  const int tid  = threadIdx.x;
  const int wave = tid >> 5;
  const int lane = tid & 31;
  const int l15  = lane & 15;
  const int half = lane >> 4;
  const int rt   = wave & 3;   // row tile within block
  const int ch   = wave >> 2;  // hidden-column half

  const int rowTileBase = blockIdx.x * 64 + rt * 16;
  const int arow  = rowTileBase + l15;
  const int arowc = arow < N ? arow : N - 1;
  const float* xrow = X + (size_t)arowc * Cin;

  const v8f vzero = {0.f, 0.f, 0.f, 0.f, 0.f, 0.f, 0.f, 0.f};
  v8f acc1[8];
#pragma unroll
  for (int t = 0; t < 8; ++t) acc1[t] = vzero;

  // -------- stage 1: H = relu(X @ W0 + b0), 16x16x32 bf16 WMMA --------
  for (int kc = 0; kc < Cin; kc += 32) {
    // Prefetch A tile (raw fp32) before barriers: overlap HBM with B staging.
    const float4* p0 = (const float4*)(xrow + kc + 8 * half);
    const float4 f0 = p0[0], f1 = p0[1];
    const float4* p1 = (const float4*)(xrow + kc + 16 + 8 * half);
    const float4 f2 = p1[0], f3 = p1[1];

    __syncthreads();
    {  // cooperative stage of W0T k-chunk: thread tid copies hidden row n=tid
      const uint4* gs = (const uint4*)(W0T + (size_t)tid * Cin + kc);
      uint4* ld = (uint4*)(ldsB + tid * 32);
      ld[0] = gs[0]; ld[1] = gs[1]; ld[2] = gs[2]; ld[3] = gs[3];
    }
    __syncthreads();

    // A fragment: lane = row M, halves hold K {0..7,16..23} / {8..15,24..31}
    AB32 a;
    a.d[0] = pk2bf(f0.x, f0.y);
    a.d[1] = pk2bf(f0.z, f0.w);
    a.d[2] = pk2bf(f1.x, f1.y);
    a.d[3] = pk2bf(f1.z, f1.w);
    a.d[4] = pk2bf(f2.x, f2.y);
    a.d[5] = pk2bf(f2.z, f2.w);
    a.d[6] = pk2bf(f3.x, f3.y);
    a.d[7] = pk2bf(f3.z, f3.w);

    // B fragments: lane = col N, halves hold K {0..7,16..23} / {8..15,24..31}
    // Software pipeline: load B(t+1) before issuing WMMA(t).
    const unsigned short* bbase = ldsB + half * 8 + (ch * 128 + l15) * 32;
    AB32 bb[2];
    bb[0].q[0] = ((const uint4*)bbase)[0];
    bb[0].q[1] = ((const uint4*)bbase)[2];
#pragma unroll
    for (int t = 0; t < 8; ++t) {
      if (t < 7) {
        const uint4* bp = (const uint4*)(bbase + (t + 1) * 16 * 32);
        bb[(t + 1) & 1].q[0] = bp[0];
        bb[(t + 1) & 1].q[1] = bp[2];
      }
      acc1[t] = __builtin_amdgcn_wmma_f32_16x16x32_bf16(
          false, a.v, false, bb[t & 1].v, (short)0, acc1[t], false, false);
    }
  }
  __syncthreads();

  // bias + relu, store H bf16 to LDS row-major [64][256]
#pragma unroll
  for (int t = 0; t < 8; ++t) {
    const int ncol = ch * 128 + t * 16 + l15;
    const float bias = b0[ncol];
#pragma unroll
    for (int i = 0; i < 8; ++i) {
      float h = acc1[t][i] + bias;
      h = h > 0.f ? h : 0.f;
      const int m = rt * 16 + half * 8 + i;  // C layout: row within tile
      ldsH[m * 256 + ncol] = f2bf(h);
    }
  }
  __syncthreads();

  // -------- stage 2: out = alpha * (H @ W1 + b1); wave handles 2 col-tiles ---
  v8f acc2[2];
  acc2[0] = vzero; acc2[1] = vzero;
  const unsigned short* hb = ldsH + (rt * 16 + l15) * 256;
  const unsigned short* w1b0 = W1T + (ch * 32 + l15) * 256 + half * 8;
  const unsigned short* w1b1 = w1b0 + 16 * 256;
#pragma unroll
  for (int kc = 0; kc < 256; kc += 32) {
    AB32 a, b0f, b1f;
    a.q[0] = *(const uint4*)(hb + kc + 8 * half);
    a.q[1] = *(const uint4*)(hb + kc + 16 + 8 * half);
    b0f.q[0] = ((const uint4*)(w1b0 + kc))[0];
    b0f.q[1] = ((const uint4*)(w1b0 + kc))[2];
    b1f.q[0] = ((const uint4*)(w1b1 + kc))[0];
    b1f.q[1] = ((const uint4*)(w1b1 + kc))[2];
    acc2[0] = __builtin_amdgcn_wmma_f32_16x16x32_bf16(
        false, a.v, false, b0f.v, (short)0, acc2[0], false, false);
    acc2[1] = __builtin_amdgcn_wmma_f32_16x16x32_bf16(
        false, a.v, false, b1f.v, (short)0, acc2[1], false, false);
  }

#pragma unroll
  for (int t = 0; t < 2; ++t) {
    const int ncol = ch * 32 + t * 16 + l15;
    const float bias = b1[ncol];
#pragma unroll
    for (int i = 0; i < 8; ++i) {
      const int grow = rowTileBase + half * 8 + i;
      if (grow < N) {
        const size_t o = (size_t)grow * 64 + ncol;
        const float v = alpha * (acc2[t][i] + bias);
        out[o] = accum ? out[o] + v : v;
      }
    }
  }
}

// ---------------------------------------------------------------------------
// Propagation kernels (combined h0 = 0.5h1+0.3h2+0.2h3, propagate ONCE:
// linearity of the APPNP operator => identical result, 3x less work)
// ---------------------------------------------------------------------------
__global__ void __launch_bounds__(256) k_init(const float* __restrict__ h0,
    const float* __restrict__ norm, float* __restrict__ tmp,
    float* __restrict__ acc, int total) {
  int idx = blockIdx.x * blockDim.x + threadIdx.x;
  if (idx >= total) return;
  const float nv = norm[idx >> 6];
  tmp[idx] = h0[idx] * nv;
  acc[idx] = 0.f;
}

// 16 threads per edge, 4 channels each (full 256B row per edge, coalesced)
__global__ void __launch_bounds__(256) k_scatter(const int* __restrict__ esrc,
    const int* __restrict__ edst, const float* __restrict__ tmp,
    float* __restrict__ acc, int nedges) {
  int idx = blockIdx.x * blockDim.x + threadIdx.x;
  int e = idx >> 4;
  if (e >= nedges) return;
  int q = idx & 15;
  int s = esrc[e], d = edst[e];
  const float4 v = ((const float4*)tmp)[(size_t)s * 16 + q];
  float* ap = acc + (size_t)d * 64 + q * 4;
  atomicAdd(ap + 0, v.x);
  atomicAdd(ap + 1, v.y);
  atomicAdd(ap + 2, v.z);
  atomicAdd(ap + 3, v.w);
}

// h = (acc*norm)*0.9 + 0.1*h0 ; produce next tmp and re-zero acc, or final out
__global__ void __launch_bounds__(256) k_update(float* __restrict__ acc,
    const float* __restrict__ norm, const float* __restrict__ h0,
    float* __restrict__ tmp, float* __restrict__ out, int total, int last) {
  int idx = blockIdx.x * blockDim.x + threadIdx.x;
  if (idx >= total) return;
  const float nv = norm[idx >> 6];
  const float h = acc[idx] * nv * 0.9f + 0.1f * h0[idx];
  if (last) {
    out[idx] = h;
  } else {
    tmp[idx] = h * nv;
    acc[idx] = 0.f;
  }
}

// ---------------------------------------------------------------------------
extern "C" void kernel_launch(void* const* d_in, const int* in_sizes, int n_in,
                              void* d_out, int out_size, void* d_ws, size_t ws_size,
                              hipStream_t stream) {
  (void)in_sizes; (void)n_in; (void)out_size; (void)ws_size;
  const int N = 100000;
  const int E = 3200000;

  const float* f1   = (const float*)d_in[0];
  const float* f2   = (const float*)d_in[1];
  const float* f3   = (const float*)d_in[2];
  const float* norm = (const float*)d_in[3];
  const int*   esrc = (const int*)d_in[4];
  const int*   edst = (const int*)d_in[5];
  const float* w10 = (const float*)d_in[6];
  const float* b10 = (const float*)d_in[7];
  const float* w11 = (const float*)d_in[8];
  const float* b11 = (const float*)d_in[9];
  const float* w20 = (const float*)d_in[10];
  const float* b20 = (const float*)d_in[11];
  const float* w21 = (const float*)d_in[12];
  const float* b21 = (const float*)d_in[13];
  const float* w30 = (const float*)d_in[14];
  const float* b30 = (const float*)d_in[15];
  const float* w31 = (const float*)d_in[16];
  const float* b31 = (const float*)d_in[17];
  float* out = (float*)d_out;  // doubles as h0 for the propagation

  char* ws = (char*)d_ws;
  size_t off = 0;
  auto wsalloc = [&](size_t bytes) {
    void* p = ws + off;
    off += (bytes + 255) & ~(size_t)255;
    return p;
  };
  float* tmp = (float*)wsalloc((size_t)N * 64 * 4);
  float* acc = (float*)wsalloc((size_t)N * 64 * 4);
  unsigned short* w10T = (unsigned short*)wsalloc((size_t)256 * 512 * 2);
  unsigned short* w11T = (unsigned short*)wsalloc((size_t)64 * 256 * 2);
  unsigned short* w20T = (unsigned short*)wsalloc((size_t)256 * 256 * 2);
  unsigned short* w21T = (unsigned short*)wsalloc((size_t)64 * 256 * 2);
  unsigned short* w30T = (unsigned short*)wsalloc((size_t)256 * 128 * 2);
  unsigned short* w31T = (unsigned short*)wsalloc((size_t)64 * 256 * 2);

  auto cvt = [&](const float* s, unsigned short* dptr, int K, int Nc) {
    int tot = K * Nc;
    k_cvt_t<<<(tot + 255) / 256, 256, 0, stream>>>(s, dptr, K, Nc);
  };
  cvt(w10, w10T, 512, 256);
  cvt(w11, w11T, 256, 64);
  cvt(w20, w20T, 256, 256);
  cvt(w21, w21T, 256, 64);
  cvt(w30, w30T, 128, 256);
  cvt(w31, w31T, 256, 64);

  const int gblocks = (N + 63) / 64;
  const size_t smem = 16384 + 32768;  // 48 KB dynamic LDS
  k_mlp<<<gblocks, 256, smem, stream>>>(f1, w10T, b10, w11T, b11, out, N, 512, 0.5f, 0);
  k_mlp<<<gblocks, 256, smem, stream>>>(f2, w20T, b20, w21T, b21, out, N, 256, 0.3f, 1);
  k_mlp<<<gblocks, 256, smem, stream>>>(f3, w30T, b30, w31T, b31, out, N, 128, 0.2f, 1);

  const int total = N * 64;
  k_init<<<(total + 255) / 256, 256, 0, stream>>>(out, norm, tmp, acc, total);

  const int sthreads = E * 16;
  for (int it = 0; it < 10; ++it) {
    k_scatter<<<(sthreads + 255) / 256, 256, 0, stream>>>(esrc, edst, tmp, acc, E);
    k_update<<<(total + 255) / 256, 256, 0, stream>>>(acc, norm, out, tmp, out, total, it == 9);
  }
}
